// EarthAttention3D_63007170232404
// MI455X (gfx1250) — matrix operand order repaired and checked by reading, not compiled, not run on hardware
//
#include <hip/hip_runtime.h>

// ---------------------------------------------------------------------------
// EarthAttention3D (Pangu-Weather) for gfx1250 — bf16 WMMA pipeline.
//   B_=960 windows, L=144, C=192, HEADS=6, hd=32, nw=64 bias groups, w_wins=15
// Pipeline:
//   0. convert x -> bf16; transpose W1/W2 -> bf16; zero V pad; precompute bias
//   1. QKV GEMM (144x192 @ 192x576) -> q(scaled)/k bf16 (L,hd rows), v bf16 transposed (hd,Lpad)
//   2. fused attention per (win,head): S=QK^T (1 WMMA/tile), +bias+mask, softmax,
//      P via LDS, O=PV (5 WMMA K-chunks, K padded 144->160)
//   3. out projection (144x192 @ 192x192) + b2 -> fp32 (streamed NT)
// ---------------------------------------------------------------------------

typedef __attribute__((ext_vector_type(16))) __bf16 v16bf;
typedef __attribute__((ext_vector_type(8)))  __bf16 v8bf;
typedef __attribute__((ext_vector_type(8)))  float  v8f;

#define HEADS  6
#define LSEQ   144
#define CDIM   192
#define HD     32
#define NWG    64
#define WWINS  15
#define TOTW   960          // NWG * WWINS
#define N1     576          // 3*CDIM
#define VPAD   160          // L padded to multiple of 32 for PV K-loop

__device__ __forceinline__ unsigned short f2bf(float f) {
    unsigned u = __float_as_uint(f);
    u += 0x7FFFu + ((u >> 16) & 1u);          // round-to-nearest-even
    return (unsigned short)(u >> 16);
}

// Build a 16-half fragment from two 16B-aligned 8-half chunks.
__device__ __forceinline__ v16bf ldfrag(const unsigned short* c0, const unsigned short* c1) {
    v8bf lo = *reinterpret_cast<const v8bf*>(c0);
    v8bf hi = *reinterpret_cast<const v8bf*>(c1);
    v16bf r;
#pragma unroll
    for (int i = 0; i < 8; ++i) { r[i] = lo[i]; r[8 + i] = hi[i]; }
    return r;
}

// ---------------- prep kernels ----------------

__global__ void k_f32_to_bf16(const float* __restrict__ src,
                              unsigned short* __restrict__ dst, int n) {
    int i = blockIdx.x * blockDim.x + threadIdx.x;
    if (i < n) dst[i] = f2bf(__builtin_nontemporal_load(src + i));
}

__global__ void k_prep_w(const float* __restrict__ W1, const float* __restrict__ W2,
                         unsigned short* __restrict__ W1T, unsigned short* __restrict__ W2T) {
    int i = blockIdx.x * blockDim.x + threadIdx.x;
    if (i < N1 * CDIM) {                       // W1T[o][c] = W1[c][o]
        int o = i / CDIM, c = i % CDIM;
        W1T[i] = f2bf(W1[c * N1 + o]);
        return;
    }
    i -= N1 * CDIM;
    if (i < CDIM * CDIM) {                     // W2T[o][c] = W2[c][o]
        int o = i / CDIM, c = i % CDIM;
        W2T[i] = f2bf(W2[c * CDIM + o]);
    }
}

__global__ void k_zero_vpad(unsigned short* __restrict__ vtb) {
    int i = blockIdx.x * blockDim.x + threadIdx.x;
    const int n = TOTW * HEADS * HD * (VPAD - LSEQ);
    if (i < n) {
        int t = i >> 4, j = i & 15;            // 16 pad cols per (win,head,d) row
        vtb[t * VPAD + LSEQ + j] = 0;
    }
}

__global__ void k_bias_pre(const float* __restrict__ table, const int* __restrict__ pos,
                           float* __restrict__ bias_pre) {
    // bias_pre[((nwi*HEADS+head)*L + l)*L + m] = table[pos[l][m]*NWG*HEADS + nwi*HEADS + head]
    int i = blockIdx.x * blockDim.x + threadIdx.x;
    const int n = NWG * HEADS * LSEQ * LSEQ;
    if (i >= n) return;
    int m = i % LSEQ; int t = i / LSEQ;
    int l = t % LSEQ; t /= LSEQ;
    int head = t % HEADS; int nwi = t / HEADS;
    bias_pre[i] = table[pos[l * LSEQ + m] * (NWG * HEADS) + nwi * HEADS + head];
}

// ---------------- kernel 1: QKV projection ----------------
// grid = 960*9 blocks (win, mtile), 256 threads = 8 waves, wave owns ntile = w, w+8, ...
__global__ __launch_bounds__(256)
void k_qkv(const unsigned short* __restrict__ xb, const unsigned short* __restrict__ W1T,
           const float* __restrict__ b1,
           unsigned short* __restrict__ qb, unsigned short* __restrict__ kb,
           unsigned short* __restrict__ vtb) {
    const int blk = blockIdx.x;
    const int win = blk / 9, mtile = blk % 9;
    const int wave = threadIdx.x >> 5, lane = threadIdx.x & 31;
    const int n = lane & 15, half = lane >> 4;

    // A fragments: row m = lane%16, K split per ISA A-layout (two 16B chunks)
    const unsigned short* xrow = xb + (win * LSEQ + mtile * 16 + n) * CDIM;
    v16bf afr[6];
#pragma unroll
    for (int kt = 0; kt < 6; ++kt) {
        const unsigned short* p0 = xrow + kt * 32 + 8 * half;
        afr[kt] = ldfrag(p0, p0 + 16);
    }

    const float scale = 0.17677669529663687f;  // hd^-0.5

    for (int ntile = wave; ntile < 36; ntile += 8) {
        v8f acc = {};
        const unsigned short* wrow = W1T + (ntile * 16 + n) * CDIM;
#pragma unroll
        for (int kt = 0; kt < 6; ++kt) {
            const unsigned short* p = wrow + kt * 32 + 8 * half;
            v16bf b = ldfrag(p, p + 16);
            acc = __builtin_amdgcn_wmma_f32_16x16x32_bf16(false, afr[kt], false, b,
                                                          (short)0, acc, false, false);
        }
        const int o = ntile * 16 + n;
        const float bo = b1[o];
        // q/k/v boundaries fall on tile boundaries: branch is wave-uniform in
        // ntile; force scalar branch via readfirstlane (no EXEC-mask dance).
        const int ntu = __builtin_amdgcn_readfirstlane(ntile);
        if (ntu < 12) {                                         // Q (scaled)
            const int head = o >> 5, d = o & 31;
            unsigned short* qp = qb + ((win * HEADS + head) * LSEQ) * HD + d;
#pragma unroll
            for (int r = 0; r < 8; ++r) {
                const int l = mtile * 16 + r + 8 * half;
                qp[l * HD] = f2bf((acc[r] + bo) * scale);
            }
        } else if (ntu < 24) {                                  // K
            const int oo = o - CDIM; const int head = oo >> 5, d = oo & 31;
            unsigned short* kp = kb + ((win * HEADS + head) * LSEQ) * HD + d;
#pragma unroll
            for (int r = 0; r < 8; ++r) {
                const int l = mtile * 16 + r + 8 * half;
                kp[l * HD] = f2bf(acc[r] + bo);
            }
        } else {                                                // V transposed (d-major)
            const int oo = o - 2 * CDIM; const int head = oo >> 5, d = oo & 31;
            unsigned short* vp = vtb + ((win * HEADS + head) * HD + d) * VPAD;
#pragma unroll
            for (int r = 0; r < 8; ++r) {
                const int l = mtile * 16 + r + 8 * half;
                vp[l] = f2bf(acc[r] + bo);
            }
        }
    }
}

// ---------------- kernel 2: fused attention per (win, head) ----------------
// grid = 960*6 blocks, 288 threads = 9 waves, wave = S row-tile (mtile)
__global__ __launch_bounds__(288)
void k_attn(const unsigned short* __restrict__ qb, const unsigned short* __restrict__ kb,
            const unsigned short* __restrict__ vtb, const float* __restrict__ bias_pre,
            const float* __restrict__ mask, unsigned short* __restrict__ ob) {
    __shared__ unsigned short pls[9 * 16 * VPAD];   // 46080 B: P tiles, bf16, K padded

    const int blk = blockIdx.x;
    const int win = blk / HEADS, head = blk % HEADS;
    const int nwi = win / WWINS;
    const int mtile = threadIdx.x >> 5;
    const int lane = threadIdx.x & 31;
    const int n = lane & 15, half = lane >> 4;

    const unsigned short* qbase = qb + (size_t)(win * HEADS + head) * LSEQ * HD;
    const unsigned short* kbase = kb + (size_t)(win * HEADS + head) * LSEQ * HD;
    const unsigned short* vbase = vtb + (size_t)(win * HEADS + head) * HD * VPAD;

    // Q A-fragment (M=16 rows, K=hd=32 — exactly one WMMA K-step)
    const unsigned short* qrow = qbase + (mtile * 16 + n) * HD + 8 * half;
    const v16bf aq = ldfrag(qrow, qrow + 16);

    // S = Q K^T : 9 column tiles, one WMMA each
    v8f s[9];
#pragma unroll
    for (int nt = 0; nt < 9; ++nt) {
        const unsigned short* kp = kbase + (nt * 16 + n) * HD + 8 * half;
        v16bf bk = ldfrag(kp, kp + 16);
        v8f z = {};
        s[nt] = __builtin_amdgcn_wmma_f32_16x16x32_bf16(false, aq, false, bk,
                                                        (short)0, z, false, false);
    }

    // + earth bias (L2-resident, reused 15x) + mask (single use -> non-temporal)
    const float* biasb = bias_pre + (size_t)(nwi * HEADS + head) * LSEQ * LSEQ;
    const float* maskb = mask + (size_t)win * LSEQ * LSEQ;
#pragma unroll
    for (int nt = 0; nt < 9; ++nt) {
#pragma unroll
        for (int r = 0; r < 8; ++r) {
            const int l = mtile * 16 + r + 8 * half;
            const int m = nt * 16 + n;
            s[nt][r] += biasb[l * LSEQ + m] +
                        __builtin_nontemporal_load(maskb + l * LSEQ + m);
        }
    }

    // softmax over m=144 : row l = r + 8*half lives in one 16-lane half-group
    float inv[8];
#pragma unroll
    for (int r = 0; r < 8; ++r) {
        float mx = s[0][r];
#pragma unroll
        for (int nt = 1; nt < 9; ++nt) mx = fmaxf(mx, s[nt][r]);
#pragma unroll
        for (int off = 1; off < 16; off <<= 1) mx = fmaxf(mx, __shfl_xor(mx, off, 32));
        float sum = 0.f;
#pragma unroll
        for (int nt = 0; nt < 9; ++nt) { float p = __expf(s[nt][r] - mx); s[nt][r] = p; sum += p; }
#pragma unroll
        for (int off = 1; off < 16; off <<= 1) sum += __shfl_xor(sum, off, 32);
        inv[r] = 1.0f / sum;
    }

    // stage P (bf16) through LDS to switch C-layout -> A-layout; zero the K pad
    unsigned short* pbase = pls + mtile * 16 * VPAD;
#pragma unroll
    for (int r = 0; r < 8; ++r) {
        const int row = r + 8 * half;
#pragma unroll
        for (int nt = 0; nt < 9; ++nt) pbase[row * VPAD + nt * 16 + n] = f2bf(s[nt][r]);
        pbase[row * VPAD + LSEQ + n] = 0;
    }
    __syncthreads();

    // O = P @ V : 2 output d-tiles, 5 K=32 chunks (144 padded to 160)
    const unsigned short* prow = pbase + n * VPAD;   // A row m = lane%16
#pragma unroll
    for (int dt = 0; dt < 2; ++dt) {
        v8f acc = {};
#pragma unroll
        for (int c = 0; c < 5; ++c) {
            const unsigned short* pa = prow + c * 32 + 8 * half;
            v16bf ap = ldfrag(pa, pa + 16);
            const unsigned short* vp = vbase + (dt * 16 + n) * VPAD + c * 32 + 8 * half;
            v16bf bv = ldfrag(vp, vp + 16);
            acc = __builtin_amdgcn_wmma_f32_16x16x32_bf16(false, ap, false, bv,
                                                          (short)0, acc, false, false);
        }
#pragma unroll
        for (int r = 0; r < 8; ++r) {
            const int l = mtile * 16 + r + 8 * half;
            ob[(size_t)(win * LSEQ + l) * CDIM + head * HD + dt * 16 + n] = f2bf(acc[r] * inv[r]);
        }
    }
}

// ---------------- kernel 3: output projection ----------------
// grid = 960*9 blocks (win, mtile), 128 threads = 4 waves, wave owns ntile = w, w+4, w+8
__global__ __launch_bounds__(128)
void k_proj(const unsigned short* __restrict__ ob, const unsigned short* __restrict__ W2T,
            const float* __restrict__ b2, float* __restrict__ out) {
    const int blk = blockIdx.x;
    const int win = blk / 9, mtile = blk % 9;
    const int wave = threadIdx.x >> 5, lane = threadIdx.x & 31;
    const int n = lane & 15, half = lane >> 4;

    const unsigned short* orow = ob + (size_t)(win * LSEQ + mtile * 16 + n) * CDIM;
    v16bf afr[6];
#pragma unroll
    for (int kt = 0; kt < 6; ++kt) {
        const unsigned short* p = orow + kt * 32 + 8 * half;
        afr[kt] = ldfrag(p, p + 16);
    }

    for (int ntile = wave; ntile < 12; ntile += 4) {
        v8f acc = {};
        const unsigned short* wrow = W2T + (ntile * 16 + n) * CDIM;
#pragma unroll
        for (int kt = 0; kt < 6; ++kt) {
            const unsigned short* p = wrow + kt * 32 + 8 * half;
            v16bf b = ldfrag(p, p + 16);
            acc = __builtin_amdgcn_wmma_f32_16x16x32_bf16(false, afr[kt], false, b,
                                                          (short)0, acc, false, false);
        }
        const int o = ntile * 16 + n;
        const float bo = b2[o];
#pragma unroll
        for (int r = 0; r < 8; ++r) {
            const int l = mtile * 16 + r + 8 * half;
            __builtin_nontemporal_store(acc[r] + bo,
                                        out + (size_t)(win * LSEQ + l) * CDIM + o);
        }
    }
}

// ---------------- host launcher ----------------

extern "C" void kernel_launch(void* const* d_in, const int* in_sizes, int n_in,
                              void* d_out, int out_size, void* d_ws, size_t ws_size,
                              hipStream_t stream) {
    const float* x          = (const float*)d_in[0];
    const float* mask       = (const float*)d_in[1];
    const float* W1         = (const float*)d_in[2];
    const float* b1         = (const float*)d_in[3];
    const float* W2         = (const float*)d_in[4];
    const float* b2         = (const float*)d_in[5];
    const float* bias_table = (const float*)d_in[6];
    const int*   pos        = (const int*)d_in[7];
    // d_in[8..10] = b, z, h scalars (implied by constants here)

    // workspace carve-up (all element counts divisible by 8 -> 16B alignment kept)
    const int XN  = TOTW * LSEQ * CDIM;            // 26,542,080
    const int QN  = TOTW * HEADS * LSEQ * HD;      // 26,542,080
    const int VN  = TOTW * HEADS * HD * VPAD;      // 29,491,200
    const int BN  = NWG * HEADS * LSEQ * LSEQ;     //  7,962,624 floats

    unsigned short* xb  = (unsigned short*)d_ws;   // also reused as ob after k_qkv
    unsigned short* w1t = xb  + XN;
    unsigned short* w2t = w1t + N1 * CDIM;
    unsigned short* qb  = w2t + CDIM * CDIM;
    unsigned short* kb  = qb  + QN;
    unsigned short* vtb = kb  + QN;
    float* bias_pre     = (float*)(vtb + VN);

    k_f32_to_bf16<<<XN / 256, 256, 0, stream>>>(x, xb, XN);
    k_prep_w<<<(N1 * CDIM + CDIM * CDIM) / 256, 256, 0, stream>>>(W1, W2, w1t, w2t);
    k_zero_vpad<<<(TOTW * HEADS * HD * (VPAD - LSEQ)) / 256, 256, 0, stream>>>(vtb);
    k_bias_pre<<<BN / 256, 256, 0, stream>>>(bias_table, pos, bias_pre);

    k_qkv<<<TOTW * 9, 256, 0, stream>>>(xb, w1t, b1, qb, kb, vtb);
    k_attn<<<TOTW * HEADS, 288, 0, stream>>>(qb, kb, vtb, bias_pre, mask, /*ob=*/xb);
    k_proj<<<TOTW * 9, 128, 0, stream>>>(/*ob=*/xb, w2t, b2, (float*)d_out);
}
